// CP_Attention_Skill_1563368096227
// MI455X (gfx1250) — matrix operand order repaired and checked by reading, not compiled, not run on hardware
//
#include <hip/hip_runtime.h>
#include <hip/hip_bf16.h>
#include <stdint.h>

// ---------------- CDNA5 WMMA types ----------------
typedef __attribute__((ext_vector_type(16))) __bf16 v16bf;
typedef __attribute__((ext_vector_type(8)))  float  v8f;

#define NHEAD 8
#define HDIM  64
#define DMODEL 512
#define NSEQ  1024
#define BATCH 8
#define ATT_SCALE 0.125f   // 64^-0.5

// ---------------- async global->LDS copy (CDNA5, ASYNCcnt) ----------------
#if defined(__gfx1250__) && __has_builtin(__builtin_amdgcn_global_load_async_to_lds_b128)
#define HAVE_ASYNC_COPY 1
typedef int v4i __attribute__((vector_size(16)));
typedef __attribute__((address_space(1))) v4i g_v4i;   // global
typedef __attribute__((address_space(3))) v4i l_v4i;   // LDS
__device__ __forceinline__ void async_copy_b128(const void* g, void* l) {
  __builtin_amdgcn_global_load_async_to_lds_b128(
      (g_v4i*)const_cast<void*>(g), (l_v4i*)l, 0, 0);
}
__device__ __forceinline__ void async_wait0() {
#if __has_builtin(__builtin_amdgcn_s_wait_asynccnt)
  __builtin_amdgcn_s_wait_asynccnt(0);
#else
  asm volatile("s_wait_asynccnt 0" ::: "memory");
#endif
}
#else
#define HAVE_ASYNC_COPY 0
#endif

// ---------------- bf16 helpers: native fptrunc (RNE) ----------------
__device__ __forceinline__ unsigned short f2bf(float f) {
  union { __bf16 h; unsigned short u; } c;
  c.h = (__bf16)f;   // lowers to native bf16 cvt
  return c.u;
}

union BF16Frag { v16bf v; unsigned short u[16]; };

// A fragment 16x32 bf16 from row-major LDS tile (stride in elements).
// Lane l: row m = l&15; K-half = (l>>4)*8. Per-lane data is two contiguous
// 16-byte runs -> 2x ds_load_b128.
__device__ __forceinline__ v16bf load_a_frag(const unsigned short* base, int stride) {
  int lane = threadIdx.x & 31;
  int m = lane & 15;
  int halfk = (lane >> 4) * 8;
  const unsigned short* row = base + m * stride;
  BF16Frag f;
#pragma unroll
  for (int i = 0; i < 8; ++i) {
    int k = ((i < 4) ? 0 : 16) + halfk + (i & 3) * 2;
    f.u[2 * i]     = row[k];
    f.u[2 * i + 1] = row[k + 1];
  }
  return f.v;
}

// B fragment 32x16 bf16 (KxN) where LDS holds the tile N-major:
// element B[k][n] = base[n*stride + k]. Lane l: n = l&15, K-half = (l>>4)*8.
// Per-lane data is two contiguous 8-bf16 runs -> 2x ds_load_b128.
__device__ __forceinline__ v16bf load_bT_frag(const unsigned short* base, int stride) {
  int lane = threadIdx.x & 31;
  int n = lane & 15;
  int kh = (lane >> 4) * 8;
  const unsigned short* row = base + (size_t)n * stride + kh;
  BF16Frag f;
#pragma unroll
  for (int i = 0; i < 8; ++i) {
    f.u[2 * i]     = row[((i < 4) ? 0 : 16) + (i & 3) * 2];
    f.u[2 * i + 1] = row[((i < 4) ? 0 : 16) + (i & 3) * 2 + 1];
  }
  return f.v;
}

// ---------------- prep kernels (tiny, VALU) ----------------
__global__ void prep_cpc(const float* __restrict__ CP_C, const float* __restrict__ cp_att,
                         float* __restrict__ cpf) {
  int idx = blockIdx.x * blockDim.x + threadIdx.x;
  if (idx >= 4 * 64 * 64) return;
  int f = idx >> 12, u = (idx >> 6) & 63, v = idx & 63;
  float s = 0.f;
#pragma unroll 8
  for (int r = 0; r < 64; ++r) s += CP_C[(size_t)u * 4096 + v * 64 + r] * cp_att[r * 4 + f];
  cpf[idx] = s;
}

__global__ void prep_T(const float* __restrict__ Wu, const float* __restrict__ cpf,
                       float* __restrict__ T) {
  int idx = blockIdx.x * blockDim.x + threadIdx.x;
  if (idx >= 4 * 512 * 64) return;
  int f = idx >> 15, i = (idx >> 6) & 511, t = idx & 63;
  float s = 0.f;
#pragma unroll 8
  for (int u = 0; u < 64; ++u) s += Wu[(size_t)i * 64 + u] * cpf[(size_t)f * 4096 + u * 64 + t];
  T[idx] = s;
}

__global__ void prep_weff(const float* __restrict__ Wq, const float* __restrict__ Wk,
                          const float* __restrict__ Wv, const float* __restrict__ Wp,
                          const float* __restrict__ T, const float* __restrict__ Wvlin,
                          unsigned short* __restrict__ Weff) {
  int idx = blockIdx.x * blockDim.x + threadIdx.x;
  if (idx >= 4 * 512 * 512) return;
  int f = idx >> 18, i = (idx >> 9) & 511, j = idx & 511;
  const float* W = (f == 0) ? Wq : (f == 1) ? Wk : (f == 2) ? Wv : Wp;
  float s = W[(size_t)i * 512 + j];
  const float* Tf = T + ((size_t)f << 15) + (size_t)i * 64;
#pragma unroll 8
  for (int t = 0; t < 64; ++t) s += Tf[t] * Wvlin[(size_t)t * 512 + j];
  Weff[idx] = f2bf(s);
}

// ---------------- main GEMM: C(MxN) = A(MxK) @ Weff(KxN) ----------------
#define GTM 128
#define GTN 64
#define GTK 32

__global__ __launch_bounds__(256)
void gemm_weff_kernel(const float* __restrict__ Af, const unsigned short* __restrict__ Ab,
                      int a_is_f32, const unsigned short* __restrict__ Bw, int ldb,
                      float* __restrict__ Cf, unsigned short* __restrict__ Cb,
                      const float* __restrict__ bias, int M, int N, int K) {
  __shared__ unsigned short As[GTM][GTK];        // row-major A tile, 8 KB
  __shared__ unsigned short Bst[GTN][GTK + 8];   // N-major (transposed) B tile, 5 KB

  int tid = threadIdx.x;
  int wave = tid >> 5, lane = tid & 31;
  int m0 = blockIdx.y * GTM;
  int n0 = blockIdx.x * GTN;

  v8f acc[4] = {};

  int ar = tid >> 1, ac = (tid & 1) * 16;   // A stage: 128 rows x (2x16)
  int br = tid >> 3, bc = (tid & 7) * 8;    // B stage: 32 k-rows x (8x8 n)

  if (a_is_f32) {
    for (int k0 = 0; k0 < K; k0 += GTK) {
      unsigned short abuf[16];
      const float* src = Af + (size_t)(m0 + ar) * K + k0 + ac;
#pragma unroll
      for (int i = 0; i < 16; ++i) abuf[i] = f2bf(src[i]);
      // RT-speculative prefetch: silently dropped if past the allocation
      __builtin_prefetch(src + GTK, 0, 1);   // global_prefetch_b8
      unsigned short bbuf[8];
      const unsigned short* bsrc = Bw + (size_t)(k0 + br) * ldb + n0 + bc;
#pragma unroll
      for (int i = 0; i < 8; ++i) bbuf[i] = bsrc[i];
      __builtin_prefetch(bsrc + (size_t)GTK * ldb, 0, 1);

      __syncthreads();  // previous iteration's fragment reads done
#pragma unroll
      for (int i = 0; i < 16; ++i) As[ar][ac + i] = abuf[i];
#pragma unroll
      for (int i = 0; i < 8; ++i) Bst[bc + i][br] = bbuf[i];   // transpose into LDS
      __syncthreads();

      v16bf afrag = load_a_frag(&As[wave * 16][0], GTK);
      v16bf bf[4];
#pragma unroll
      for (int j = 0; j < 4; ++j) bf[j] = load_bT_frag(&Bst[j * 16][0], GTK + 8);
#pragma unroll
      for (int j = 0; j < 4; ++j)
        acc[j] = __builtin_amdgcn_wmma_f32_16x16x32_bf16(
            false, afrag, false, bf[j], (short)0, acc[j], false, false);
    }
  } else {
    for (int k0 = 0; k0 < K; k0 += GTK) {
      const unsigned short* asrc = Ab + (size_t)(m0 + ar) * K + k0 + ac;
      unsigned short bbuf[8];
      const unsigned short* bsrc = Bw + (size_t)(k0 + br) * ldb + n0 + bc;
#pragma unroll
      for (int i = 0; i < 8; ++i) bbuf[i] = bsrc[i];

      __syncthreads();
#if HAVE_ASYNC_COPY
      async_copy_b128(asrc,     &As[ar][ac]);      // bf16 A: pure copy -> async DMA
      async_copy_b128(asrc + 8, &As[ar][ac + 8]);
#else
#pragma unroll
      for (int i = 0; i < 16; ++i) As[ar][ac + i] = asrc[i];
#endif
#pragma unroll
      for (int i = 0; i < 8; ++i) Bst[bc + i][br] = bbuf[i];
#if HAVE_ASYNC_COPY
      async_wait0();   // our async LDS writes landed before we signal the barrier
#endif
      __syncthreads();

      v16bf afrag = load_a_frag(&As[wave * 16][0], GTK);
      v16bf bf[4];
#pragma unroll
      for (int j = 0; j < 4; ++j) bf[j] = load_bT_frag(&Bst[j * 16][0], GTK + 8);
#pragma unroll
      for (int j = 0; j < 4; ++j)
        acc[j] = __builtin_amdgcn_wmma_f32_16x16x32_bf16(
            false, afrag, false, bf[j], (short)0, acc[j], false, false);
    }
  }

  int hh = lane >> 4, nn = lane & 15;
#pragma unroll
  for (int j = 0; j < 4; ++j) {
#pragma unroll
    for (int i = 0; i < 8; ++i) {
      int row = m0 + wave * 16 + i + 8 * hh;
      int col = n0 + j * 16 + nn;
      float v = acc[j][i];
      if (bias) v += bias[col];
      if (Cf) Cf[(size_t)row * N + col] = v;
      else    Cb[(size_t)row * N + col] = f2bf(v);
    }
  }
}

// ---------------- flash attention, one (b,h) x 128 query rows per block ----------------
__global__ __launch_bounds__(256)
void attn_kernel(const unsigned short* __restrict__ qb, const unsigned short* __restrict__ kb,
                 const unsigned short* __restrict__ vb, const unsigned char* __restrict__ mask,
                 unsigned short* __restrict__ xb) {
  int bh = blockIdx.y;
  int b = bh / NHEAD, h = bh % NHEAD;
  int q0 = blockIdx.x * 128;
  int tid = threadIdx.x, wave = tid >> 5, lane = tid & 31;

  __shared__ unsigned short Qs[128][HDIM + 8];     // row-major Q tile (row stride 144B)
  __shared__ unsigned short Ks[64][HDIM + 8];      // row-major K chunk
  __shared__ unsigned short Vst[HDIM][64 + 8];     // d-major (transposed) V chunk
  __shared__ unsigned short Ps[8][16][HDIM + 8];   // per-wave P scratch (row-major)

  // Stage Q tile (128 x 64 bf16): pure copy -> async DMA to LDS
  {
    int r = tid >> 1, c = (tid & 1) * 32;
    const unsigned short* src =
        qb + ((size_t)b * NSEQ + q0 + r) * (NHEAD * HDIM) + h * HDIM + c;
#if HAVE_ASYNC_COPY
#pragma unroll
    for (int i = 0; i < 4; ++i) async_copy_b128(src + 8 * i, &Qs[r][c + 8 * i]);
    async_wait0();
#else
#pragma unroll
    for (int i = 0; i < 32; ++i) Qs[r][c + i] = src[i];
#endif
  }
  __syncthreads();
  v16bf qfrag[2];
  qfrag[0] = load_a_frag(&Qs[wave * 16][0],  HDIM + 8);
  qfrag[1] = load_a_frag(&Qs[wave * 16][32], HDIM + 8);

  v8f Oacc[4] = {};
  float mrow[8], lrow[8];
#pragma unroll
  for (int i = 0; i < 8; ++i) { mrow[i] = -3.0e38f; lrow[i] = 0.f; }

  int hh = lane >> 4, nn = lane & 15;

  for (int kc = 0; kc < NSEQ; kc += 64) {
    // Stage K (row-major, async) and V (transposed, via regs) chunk, 64x64 each
    {
      int r = tid >> 2, c = (tid & 3) * 16;
      const unsigned short* ksrc =
          kb + ((size_t)b * NSEQ + kc + r) * (NHEAD * HDIM) + h * HDIM + c;
      const unsigned short* vsrc =
          vb + ((size_t)b * NSEQ + kc + r) * (NHEAD * HDIM) + h * HDIM + c;
      unsigned short vt[16];
#pragma unroll
      for (int i = 0; i < 16; ++i) vt[i] = vsrc[i];
      __syncthreads();  // previous iteration's fragment reads done
#if HAVE_ASYNC_COPY
      async_copy_b128(ksrc,     &Ks[r][c]);
      async_copy_b128(ksrc + 8, &Ks[r][c + 8]);
#else
      unsigned short kt[16];
#pragma unroll
      for (int i = 0; i < 16; ++i) kt[i] = ksrc[i];
#pragma unroll
      for (int i = 0; i < 16; ++i) Ks[r][c + i] = kt[i];
#endif
#pragma unroll
      for (int i = 0; i < 16; ++i) Vst[c + i][r] = vt[i];
#if HAVE_ASYNC_COPY
      async_wait0();
#endif
      __syncthreads();
    }

    // S = Q @ K^T  (16 x 64 per wave); B[k][n] = Ks[key n][d k] -> row-contiguous.
    // Group the 4 fragment loads per k-step, then 4 independent WMMAs back-to-back.
    v8f sacc[4] = {};
#pragma unroll
    for (int ks = 0; ks < 2; ++ks) {
      v16bf bf[4];
#pragma unroll
      for (int j = 0; j < 4; ++j) bf[j] = load_bT_frag(&Ks[j * 16][ks * 32], HDIM + 8);
#pragma unroll
      for (int j = 0; j < 4; ++j)
        sacc[j] = __builtin_amdgcn_wmma_f32_16x16x32_bf16(
            false, qfrag[ks], false, bf[j], (short)0, sacc[j], false, false);
    }

    // scale + mask (mask column-uniform per lane)
#pragma unroll
    for (int j = 0; j < 4; ++j) {
      int key = kc + j * 16 + nn;
      float mv = mask[(size_t)b * NSEQ + key] ? 0.f : -3.0e38f;
#pragma unroll
      for (int i = 0; i < 8; ++i) sacc[j][i] = sacc[j][i] * ATT_SCALE + mv;
    }

    // online softmax: row max / correction / row sum (reduce within 16-lane halves)
    float corr[8];
#pragma unroll
    for (int i = 0; i < 8; ++i) {
      float v = fmaxf(fmaxf(sacc[0][i], sacc[1][i]), fmaxf(sacc[2][i], sacc[3][i]));
      v = fmaxf(v, __shfl_xor(v, 8, 32));
      v = fmaxf(v, __shfl_xor(v, 4, 32));
      v = fmaxf(v, __shfl_xor(v, 2, 32));
      v = fmaxf(v, __shfl_xor(v, 1, 32));
      float mn = fmaxf(mrow[i], v);
      corr[i] = __expf(mrow[i] - mn);
      mrow[i] = mn;
    }

    float rsum[8];
#pragma unroll
    for (int i = 0; i < 8; ++i) rsum[i] = 0.f;
#pragma unroll
    for (int j = 0; j < 4; ++j) {
#pragma unroll
      for (int i = 0; i < 8; ++i) {
        float p = __expf(sacc[j][i] - mrow[i]);
        rsum[i] += p;
        Ps[wave][i + 8 * hh][j * 16 + nn] = f2bf(p);  // same-wave LDS scratch
      }
    }
#pragma unroll
    for (int i = 0; i < 8; ++i) {
      float v = rsum[i];
      v += __shfl_xor(v, 8, 32);
      v += __shfl_xor(v, 4, 32);
      v += __shfl_xor(v, 2, 32);
      v += __shfl_xor(v, 1, 32);
      lrow[i] = lrow[i] * corr[i] + v;
#pragma unroll
      for (int j = 0; j < 4; ++j) Oacc[j][i] *= corr[i];
    }

    // O += P @ V  : B[k=key][n=d] = Vst[d][key] -> row-contiguous fragment loads
    v16bf pfrag[2];
    pfrag[0] = load_a_frag(&Ps[wave][0][0],  HDIM + 8);
    pfrag[1] = load_a_frag(&Ps[wave][0][32], HDIM + 8);
#pragma unroll
    for (int ks = 0; ks < 2; ++ks) {
      v16bf bf[4];
#pragma unroll
      for (int j = 0; j < 4; ++j) bf[j] = load_bT_frag(&Vst[j * 16][ks * 32], 64 + 8);
#pragma unroll
      for (int j = 0; j < 4; ++j)
        Oacc[j] = __builtin_amdgcn_wmma_f32_16x16x32_bf16(
            false, pfrag[ks], false, bf[j], (short)0, Oacc[j], false, false);
    }
  }

  // epilogue: normalize and write x in (B, N, H*HD) layout
#pragma unroll
  for (int i = 0; i < 8; ++i) {
    float rl = (lrow[i] > 0.f) ? (1.0f / lrow[i]) : 0.f;
    int row = q0 + wave * 16 + i + 8 * hh;
#pragma unroll
    for (int j = 0; j < 4; ++j) {
      int col = j * 16 + nn;
      xb[((size_t)b * NSEQ + row) * (NHEAD * HDIM) + h * HDIM + col] = f2bf(Oacc[j][i] * rl);
    }
  }
}

// ---------------- launch ----------------
extern "C" void kernel_launch(void* const* d_in, const int* in_sizes, int n_in,
                              void* d_out, int out_size, void* d_ws, size_t ws_size,
                              hipStream_t stream) {
  (void)in_sizes; (void)n_in; (void)out_size; (void)ws_size;
  const float*        input_q = (const float*)d_in[0];
  const float*        input_k = (const float*)d_in[1];
  const float*        input_v = (const float*)d_in[2];
  const unsigned char* mask   = (const unsigned char*)d_in[3];
  const float* Wq     = (const float*)d_in[4];
  const float* Wk     = (const float*)d_in[5];
  const float* Wval   = (const float*)d_in[6];
  const float* Wp     = (const float*)d_in[7];
  const float* bp     = (const float*)d_in[8];
  const float* cp_att = (const float*)d_in[9];
  const float* CP_C   = (const float*)d_in[10];
  const float* Wu     = (const float*)d_in[11];
  const float* Wvlin  = (const float*)d_in[12];
  float* out = (float*)d_out;

  const size_t M = (size_t)BATCH * NSEQ;  // 8192

  // workspace layout
  float* cpf = (float*)d_ws;                                   // 4*64*64 f32
  float* T   = cpf + 4 * 64 * 64;                              // 4*512*64 f32
  unsigned short* Weff = (unsigned short*)(T + 4 * 512 * 64);  // 4*512*512 bf16
  unsigned short* qb = Weff + (size_t)4 * 512 * 512;           // M*512 bf16 each
  unsigned short* kb = qb + M * 512;
  unsigned short* vb = kb + M * 512;
  unsigned short* xb = vb + M * 512;

  // prep: fold CP adapters into effective weights
  prep_cpc <<<(4 * 64 * 64 + 255) / 256, 256, 0, stream>>>(CP_C, cp_att, cpf);
  prep_T   <<<(4 * 512 * 64) / 256,      256, 0, stream>>>(Wu, cpf, T);
  prep_weff<<<(4 * 512 * 512) / 256,     256, 0, stream>>>(Wq, Wk, Wval, Wp, T, Wvlin, Weff);

  // q/k/v projections: bf16 WMMA GEMMs, outputs bf16
  dim3 g(DMODEL / GTN, (unsigned)(M / GTM));  // (8, 64)
  gemm_weff_kernel<<<g, 256, 0, stream>>>(input_q, nullptr, 1, Weff,                         DMODEL,
                                          nullptr, qb, nullptr, (int)M, DMODEL, DMODEL);
  gemm_weff_kernel<<<g, 256, 0, stream>>>(input_k, nullptr, 1, Weff + (size_t)1 * 512 * 512, DMODEL,
                                          nullptr, kb, nullptr, (int)M, DMODEL, DMODEL);
  gemm_weff_kernel<<<g, 256, 0, stream>>>(input_v, nullptr, 1, Weff + (size_t)2 * 512 * 512, DMODEL,
                                          nullptr, vb, nullptr, (int)M, DMODEL, DMODEL);

  // flash attention
  dim3 ga(NSEQ / 128, BATCH * NHEAD);  // (8, 64)
  attn_kernel<<<ga, 256, 0, stream>>>(qb, kb, vb, mask, xb);

  // output projection (+bias), fp32 out
  gemm_weff_kernel<<<g, 256, 0, stream>>>(nullptr, xb, 0, Weff + (size_t)3 * 512 * 512, DMODEL,
                                          out, nullptr, bp, (int)M, DMODEL, DMODEL);
}
